// BasicBlock_XNOR_1_51092930953594
// MI455X (gfx1250) — 8-bit matrix operand order repaired and checked by reading, compiled, not run on hardware
//
#include <hip/hip_runtime.h>
#include <hip/hip_bf16.h>

typedef __attribute__((ext_vector_type(8))) int v8i;

#define NB   16
#define CIN  256
#define PP   256
#define PH   128
#define HH   56
#define WW   56
#define HWSZ (HH * WW)

__global__ void k_bn_stats(const float* __restrict__ x, const float* __restrict__ gamma,
                           const float* __restrict__ beta, const float* __restrict__ mv,
                           float* __restrict__ scale, float* __restrict__ shift,
                           int C)
{
    __shared__ float s1[256], s2[256];
    const int c = blockIdx.x;
    float a = 0.f, b = 0.f;
    const long cnt = (long)NB * HWSZ;
    for (long i = threadIdx.x; i < cnt; i += blockDim.x) {
        long n = i / HWSZ, r = i - n * HWSZ;
        float v = x[(n * C + c) * HWSZ + r];
        a += v; b += v * v;
    }
    s1[threadIdx.x] = a; s2[threadIdx.x] = b;
    __syncthreads();
    for (int st = 128; st > 0; st >>= 1) {
        if ((int)threadIdx.x < st) {
            s1[threadIdx.x] += s1[threadIdx.x + st];
            s2[threadIdx.x] += s2[threadIdx.x + st];
        }
        __syncthreads();
    }
    if (threadIdx.x == 0) {
        float mean = s1[0] / (float)cnt;
        float var  = s2[0] / (float)cnt - mean * mean;
        float sc   = gamma[c] * rsqrtf(var + 1e-5f);
        scale[c] = sc;
        shift[c] = beta[c] - mean * sc + mv[c];
    }
}

__global__ void k_binarize(const float* __restrict__ x, const float* __restrict__ scale,
                           const float* __restrict__ shift, signed char* __restrict__ out,
                           int C, long total)
{
    long tid = (long)blockIdx.x * blockDim.x + threadIdx.x;
    if (tid >= total) return;
    int c = (int)(tid % C); long r = tid / C;
    int w = (int)(r % WW); r /= WW;
    int h = (int)(r % HH); r /= HH;
    int n = (int)r;
    float v = scale[c] * x[(((long)n * C + c) * HH + h) * WW + w] + shift[c];
    out[tid] = (signed char)((v > 0.f) - (v < 0.f));
}

__global__ void k_wsf(const float* __restrict__ w, float* __restrict__ sf, int K)
{
    __shared__ float sm[256];
    const int o = blockIdx.x;
    const float* wo = w + (long)o * K;
    float s = 0.f;
    for (int i = threadIdx.x; i < K; i += blockDim.x) s += fabsf(wo[i]);
    sm[threadIdx.x] = s;
    __syncthreads();
    for (int st = 128; st > 0; st >>= 1) {
        if ((int)threadIdx.x < st) sm[threadIdx.x] += sm[threadIdx.x + st];
        __syncthreads();
    }
    if (threadIdx.x == 0) sf[o] = sm[0] / (float)K;
}

__global__ void k_pack_w(const float* __restrict__ w, signed char* __restrict__ wpk,
                         int G, int OPG, int OTPG, int KCH, int CCPT, int Ig, long total)
{
    long tid = (long)blockIdx.x * blockDim.x + threadIdx.x;
    if (tid >= total) return;
    int j    = (int)(tid & 31);
    long r   = tid >> 5;
    int lane = (int)(r & 31); r >>= 5;
    int kc   = (int)(r % KCH); r /= KCH;
    int ot   = (int)(r % OTPG); r /= OTPG;
    int g    = (int)r;

    int v  = j >> 2, bj = j & 3;
    int K  = (v & 1) * 4 + bj + (lane >> 4) * 8 + (v >> 1) * 16;
    int t  = kc / CCPT;
    int cc = kc - t * CCPT;
    int ch = cc * 64 + K;
    int o  = g * OPG + ot * 16 + (lane & 15);
    float wv = w[((long)o * Ig + ch) * 9 + t];
    wpk[tid] = (signed char)((wv > 0.f) - (wv < 0.f));
}

template <int CCPT>
__global__ void __launch_bounds__(128)
k_bconv_wmma(const signed char* __restrict__ act,
             const signed char* __restrict__ wpk,
             const float* __restrict__ sf,
             const float* __restrict__ alpha,
             float* __restrict__ out,
             int Cact, int Cg, int Cout, int OPG, int OTPG)
{
    const int WT   = (WW + 15) >> 4;
    const int lane = threadIdx.x & 31;
    const int wid  = threadIdx.x >> 5;

    int tile = blockIdx.x;
    const int wt = tile % WT; tile /= WT;
    const int h  = tile % HH; tile /= HH;
    const int n  = tile;

    const int otf = blockIdx.y * 4 + wid;
    const int g   = otf / OTPG;
    const int ot  = otf - g * OTPG;

    const int w0   = wt << 4;
    const int m    = lane & 15;
    const int half = lane >> 4;
    const int wpix = w0 + m;

    const int KCH = 9 * CCPT;
    const signed char* wslot = wpk + ((long)(g * OTPG + ot) * KCH) * 1024 + (long)lane * 32;

    v8i acc = {0, 0, 0, 0, 0, 0, 0, 0};

    #pragma unroll
    for (int t = 0; t < 9; ++t) {
        const int kh = t / 3 - 1, kw = t % 3 - 1;
        const int hi = h + kh, wi = wpix + kw;
        const bool valid = (hi >= 0) && (hi < HH) && (wi >= 0) && (wi < WW);
        const long pixBase = ((long)(n * HH + hi) * WW + wi) * Cact + g * Cg;
        #pragma unroll
        for (int cc = 0; cc < CCPT; ++cc) {
            v8i a;
            if (valid) {
                const int* ap = (const int*)(act + pixBase + cc * 64 + half * 8);
                a[0] = ap[0];  a[1] = ap[1];
                a[2] = ap[4];  a[3] = ap[5];
                a[4] = ap[8];  a[5] = ap[9];
                a[6] = ap[12]; a[7] = ap[13];
            } else {
                a = (v8i){0, 0, 0, 0, 0, 0, 0, 0};
            }
            const int* bp = (const int*)(wslot + (long)(t * CCPT + cc) * 1024);
            v8i b;
            b[0] = bp[0]; b[1] = bp[1]; b[2] = bp[2]; b[3] = bp[3];
            b[4] = bp[4]; b[5] = bp[5]; b[6] = bp[6]; b[7] = bp[7];
            acc = __builtin_amdgcn_wmma_i32_16x16x64_iu8(true, a, true, b, acc,
                                                         false, false);
        }
    }

    const int o  = g * OPG + ot * 16 + (lane & 15);
    const float s  = sf[o];
    const float al = alpha[o];
    float* orow = out + (((long)n * Cout + o) * HH + h) * WW;
    #pragma unroll
    for (int r = 0; r < 8; ++r) {
        const int wo = w0 + r + half * 8;
        if (wo < WW) {
            float v = s * (float)acc[r];
            orow[wo] = (v >= 0.f) ? v : al * v;
        }
    }
}

__global__ void k_final(const float* __restrict__ y, const float* __restrict__ x,
                        const float* __restrict__ scale, const float* __restrict__ shift,
                        const float* __restrict__ a3, const float* __restrict__ m4,
                        float* __restrict__ out, long total)
{
    long tid = (long)blockIdx.x * blockDim.x + threadIdx.x;
    if (tid >= total) return;
    int c = (int)((tid / HWSZ) % PP);
    float v = scale[c] * y[tid] + shift[c] + x[tid];
    v = (v >= 0.f) ? v : a3[c] * v;
    out[tid] = v + m4[c];
}

extern "C" void kernel_launch(void* const* d_in, const int* in_sizes, int n_in,
                              void* d_out, int out_size, void* d_ws, size_t ws_size,
                              hipStream_t stream)
{
    const float* x  = (const float*)d_in[0];
    const float* w1 = (const float*)d_in[1];
    const float* w2 = (const float*)d_in[2];
    const float* g1 = (const float*)d_in[3];
    const float* b1 = (const float*)d_in[4];
    const float* g2 = (const float*)d_in[5];
    const float* b2 = (const float*)d_in[6];
    const float* g3 = (const float*)d_in[7];
    const float* b3 = (const float*)d_in[8];
    const float* m1 = (const float*)d_in[9];
    const float* m2 = (const float*)d_in[10];
    const float* m3 = (const float*)d_in[11];
    const float* m4 = (const float*)d_in[12];
    const float* pa1 = (const float*)d_in[13];
    const float* pa2 = (const float*)d_in[14];
    const float* pa3 = (const float*)d_in[15];
    float* out = (float*)d_out;

    char* ws = (char*)d_ws;
    size_t cur = 0;
    auto carve = [&](size_t bytes) -> char* {
        char* p = ws + cur;
        cur = (cur + bytes + 255) & ~(size_t)255;
        return p;
    };
    float* scale1 = (float*)carve(CIN * 4);
    float* shift1 = (float*)carve(CIN * 4);
    float* scale2 = (float*)carve(PH * 4);
    float* shift2 = (float*)carve(PH * 4);
    float* scale3 = (float*)carve(PP * 4);
    float* shift3 = (float*)carve(PP * 4);
    float* sf1    = (float*)carve(PH * 4);
    float* sf2    = (float*)carve(PP * 4);
    const long wp1Bytes = 2L * 4 * 18 * 1024;
    const long wp2Bytes = 2L * 8 * 9 * 1024;
    signed char* wp1 = (signed char*)carve(wp1Bytes);
    signed char* wp2 = (signed char*)carve(wp2Bytes);
    const long a1Elems = (long)NB * HWSZ * CIN;
    const long y1Elems = (long)NB * PH * HWSZ;
    const long a2Elems = (long)NB * HWSZ * PH;
    const long y2Elems = (long)NB * PP * HWSZ;
    signed char* a1i8 = (signed char*)carve(a1Elems);
    float*       y1   = (float*)carve(y1Elems * 4);
    signed char* a2i8 = (signed char*)carve(a2Elems);
    float*       y2   = (float*)carve(y2Elems * 4);
    (void)ws_size; (void)n_in; (void)in_sizes; (void)out_size;

    const int TPB = 256;
    const int convGridX = NB * HH * ((WW + 15) / 16);

    k_bn_stats<<<CIN, TPB, 0, stream>>>(x, g1, b1, m1, scale1, shift1, CIN);
    k_binarize<<<(int)((a1Elems + TPB - 1) / TPB), TPB, 0, stream>>>(
        x, scale1, shift1, a1i8, CIN, a1Elems);

    k_wsf<<<PH, TPB, 0, stream>>>(w1, sf1, (CIN / 2) * 9);
    k_wsf<<<PP, TPB, 0, stream>>>(w2, sf2, (PH / 2) * 9);
    k_pack_w<<<(int)((wp1Bytes + TPB - 1) / TPB), TPB, 0, stream>>>(
        w1, wp1, 2, 64, 4, 18, 2, CIN / 2, wp1Bytes);
    k_pack_w<<<(int)((wp2Bytes + TPB - 1) / TPB), TPB, 0, stream>>>(
        w2, wp2, 2, 128, 8, 9, 1, PH / 2, wp2Bytes);

    k_bconv_wmma<2><<<dim3(convGridX, 2), 128, 0, stream>>>(
        a1i8, wp1, sf1, pa1, y1, CIN, 128, PH, 64, 4);

    k_bn_stats<<<PH, TPB, 0, stream>>>(y1, g2, b2, m2, scale2, shift2, PH);
    k_binarize<<<(int)((a2Elems + TPB - 1) / TPB), TPB, 0, stream>>>(
        y1, scale2, shift2, a2i8, PH, a2Elems);

    k_bconv_wmma<1><<<dim3(convGridX, 4), 128, 0, stream>>>(
        a2i8, wp2, sf2, pa2, y2, PH, 64, PP, 128, 8);

    k_bn_stats<<<PP, TPB, 0, stream>>>(y2, g3, b3, m3, scale3, shift3, PP);
    k_final<<<(int)((y2Elems + TPB - 1) / TPB), TPB, 0, stream>>>(
        y2, x, scale3, shift3, pa3, m4, out, y2Elems);
}
